// JointMultiheadAttentionWeightedSum_90211493085426
// MI455X (gfx1250) — matrix operand order repaired and checked by reading, not compiled, not run on hardware
//
#include <hip/hip_runtime.h>
#include <hip/hip_bf16.h>
#include <math.h>

// Shapes fixed by the reference: T=2048, B=2, E=1024, H=16, D=64, M=1024.
#define E_DIM 1024
#define NHEAD 16
#define DH 64
#define NEGV (-1e9f)

typedef __attribute__((ext_vector_type(16))) __bf16 v16bf;
typedef __attribute__((ext_vector_type(8)))  __bf16 v8bf;
typedef __attribute__((ext_vector_type(8)))  float  v8f;

static __device__ __forceinline__ v8f wmma_bf16(v16bf a, v16bf b, v8f c) {
  // D = A(16x32 bf16) * B(32x16 bf16) + C(16x16 f32)
  return __builtin_amdgcn_wmma_f32_16x16x32_bf16(false, a, false, b, (short)0, c,
                                                 false, false);
}

static __device__ __forceinline__ v16bf cat8(v8bf lo, v8bf hi) {
  return __builtin_shufflevector(lo, hi, 0, 1, 2, 3, 4, 5, 6, 7,
                                 8, 9, 10, 11, 12, 13, 14, 15);
}

// ---------------------------------------------------------------------------
// f32 -> bf16 conversion (grid-stride). Inputs are read exactly once -> NT.
// ---------------------------------------------------------------------------
__global__ void cvt_f32_bf16(const float* __restrict__ in,
                             __bf16* __restrict__ out, long n) {
  long i = (long)blockIdx.x * blockDim.x + threadIdx.x;
  long stride = (long)gridDim.x * blockDim.x;
  for (; i < n; i += stride) out[i] = (__bf16)__builtin_nontemporal_load(in + i);
}

// ---------------------------------------------------------------------------
// NT GEMM: out[row, f] = scale * (sum_e A[row,e] * W[f,e] + bias[f])
// A: [rows, 1024] bf16 row-major.  W: [1024, 1024] bf16 row-major.
// outMode 0: bf16 [B,H,L,D]   (Q/K/KM)
// outMode 1: bf16 [B,H,D,L]   (V/VM transposed)
// outMode 2: f32  [rows,1024] (final projection, NT stores)
// Block = 128 threads (4 waves); each wave -> 16 rows x 64 cols.
// ---------------------------------------------------------------------------
__global__ void __launch_bounds__(128)
gemm_nt_bf16(const __bf16* __restrict__ A, const __bf16* __restrict__ W,
             const float* __restrict__ bias, void* __restrict__ out,
             int rows, int Bsz, int outMode, float scale) {
  const int lane = threadIdx.x & 31;
  const int wave = threadIdx.x >> 5;
  const int ln = lane & 15;
  const int half = lane >> 4;
  const int rowBase = blockIdx.y * 64 + wave * 16;
  const int colBase = blockIdx.x * 64;

  v8f zero = {0.f, 0.f, 0.f, 0.f, 0.f, 0.f, 0.f, 0.f};
  v8f acc[4];
#pragma unroll
  for (int nt = 0; nt < 4; ++nt) acc[nt] = zero;

  const __bf16* aRowPtr = A + (size_t)(rowBase + ln) * E_DIM;

  for (int kk = 0; kk < E_DIM; kk += 32) {
    const int kb = kk + half * 16;
    v8bf a0 = *(const v8bf*)(aRowPtr + kb);
    v8bf a1 = *(const v8bf*)(aRowPtr + kb + 8);
    v16bf af = cat8(a0, a1);
    const int kh = kk + half * 16;
#pragma unroll
    for (int nt = 0; nt < 4; ++nt) {
      const __bf16* wRowPtr =
          W + (size_t)(colBase + nt * 16 + ln) * E_DIM + kh;
      v16bf bf = *(const v16bf*)wRowPtr;
      acc[nt] = wmma_bf16(af, bf, acc[nt]);
    }
  }

  const int Ldim = rows / Bsz;
#pragma unroll
  for (int nt = 0; nt < 4; ++nt) {
    const int f = colBase + nt * 16 + ln;
    const float bv = bias[f];
    const int h = f >> 6;
    const int d = f & 63;
#pragma unroll
    for (int r = 0; r < 8; ++r) {
      const int row = rowBase + r + 8 * half;  // C layout: VGPR r -> M=r+8*half
      const float val = (acc[nt][r] + bv) * scale;
      if (outMode == 2) {
        // final output: write-once, never re-read on device -> NT store
        __builtin_nontemporal_store(val,
                                    (float*)out + (size_t)row * E_DIM + f);
      } else {
        const int t = row / Bsz;
        const int b = row % Bsz;
        size_t idx;
        if (outMode == 0)
          idx = (((size_t)(b * NHEAD + h) * Ldim) + t) * DH + d;
        else
          idx = (((size_t)(b * NHEAD + h) * DH) + d) * Ldim + t;
        ((__bf16*)out)[idx] = (__bf16)val;
      }
    }
  }
}

// ---------------------------------------------------------------------------
// One online-softmax attention pass for a 16-row query tile (single wave).
// Kb:  [Slen, 64] bf16 (keys, row-major).  VTb: [64, Slen] bf16 (values^T).
// CAUSAL adds position bias (non-temporal: 256MB single-use stream, larger
// than the 192MB L2 -- keep it from evicting reused K/V/W tiles) and masks
// s > t.
// ---------------------------------------------------------------------------
template <bool CAUSAL>
static __device__ __forceinline__ void attn_pass(
    v16bf qf0, v16bf qf1, const __bf16* __restrict__ Kb,
    const __bf16* __restrict__ VTb, int Slen, int sLimit,
    const float* __restrict__ posBase, long posStride, int t0, int lane,
    __bf16* pTile, float* oOut /*[32]*/) {
  const int ln = lane & 15;
  const int half = lane >> 4;

  float rmax[8], rsum[8];
  v8f zero = {0.f, 0.f, 0.f, 0.f, 0.f, 0.f, 0.f, 0.f};
  v8f o[4];
#pragma unroll
  for (int r = 0; r < 8; ++r) { rmax[r] = -1e30f; rsum[r] = 0.f; }
#pragma unroll
  for (int dt = 0; dt < 4; ++dt) o[dt] = zero;

  for (int s0 = 0; s0 < sLimit; s0 += 32) {
    // --- scores: Q(16x64) . K^T  -> two 16x16 tiles (cols s0..s0+31)
    v8f c0 = zero, c1 = zero;
    {
      const int kh = half * 16;
      const __bf16* k0 = Kb + (size_t)(s0 + ln) * DH;
      const __bf16* k1 = Kb + (size_t)(s0 + 16 + ln) * DH;
      c0 = wmma_bf16(qf0, *(const v16bf*)(k0 + kh), c0);
      c1 = wmma_bf16(qf0, *(const v16bf*)(k1 + kh), c1);
      c0 = wmma_bf16(qf1, *(const v16bf*)(k0 + 32 + kh), c0);
      c1 = wmma_bf16(qf1, *(const v16bf*)(k1 + 32 + kh), c1);
    }

    // --- bias + mask + online softmax (per-row stats inside the 16-lane half)
    float p0[8], p1[8], corr[8];
#pragma unroll
    for (int r = 0; r < 8; ++r) {
      const int t = t0 + r + 8 * half;
      float x0 = c0[r], x1 = c1[r];
      if (CAUSAL) {
        const float* pr = posBase + (size_t)t * posStride + s0;
        x0 += __builtin_nontemporal_load(pr + ln);
        x1 += __builtin_nontemporal_load(pr + 16 + ln);
        if (s0 + ln > t) x0 = NEGV;
        if (s0 + 16 + ln > t) x1 = NEGV;
      }
      float cm = fmaxf(x0, x1);
#pragma unroll
      for (int off = 8; off; off >>= 1) cm = fmaxf(cm, __shfl_xor(cm, off, 32));
      const float nm = fmaxf(rmax[r], cm);
      corr[r] = __expf(rmax[r] - nm);
      p0[r] = __expf(x0 - nm);
      p1[r] = __expf(x1 - nm);
      float cs = p0[r] + p1[r];
#pragma unroll
      for (int off = 8; off; off >>= 1) cs += __shfl_xor(cs, off, 32);
      rsum[r] = rsum[r] * corr[r] + cs;
      rmax[r] = nm;
    }
#pragma unroll
    for (int dt = 0; dt < 4; ++dt)
#pragma unroll
      for (int r = 0; r < 8; ++r) o[dt][r] *= corr[r];

    // --- route P(16x32) through LDS to re-fragment into WMMA A layout
#pragma unroll
    for (int r = 0; r < 8; ++r) {
      const int m = r + 8 * half;
      pTile[m * 32 + ln] = (__bf16)p0[r];
      pTile[m * 32 + 16 + ln] = (__bf16)p1[r];
    }
    __syncthreads();
    v8bf pa0 = *(const v8bf*)(pTile + ln * 32 + half * 16);
    v8bf pa1 = *(const v8bf*)(pTile + ln * 32 + half * 16 + 8);
    v16bf pf = cat8(pa0, pa1);

    // --- O += P . V   (V^T rows are contiguous in the contraction dim)
#pragma unroll
    for (int dt = 0; dt < 4; ++dt) {
      const __bf16* vrow = VTb + (size_t)(dt * 16 + ln) * Slen + s0 + half * 16;
      o[dt] = wmma_bf16(pf, *(const v16bf*)vrow, o[dt]);
    }
    __syncthreads();
  }

#pragma unroll
  for (int dt = 0; dt < 4; ++dt)
#pragma unroll
    for (int r = 0; r < 8; ++r) oOut[dt * 8 + r] = o[dt][r] / rsum[r];
}

// ---------------------------------------------------------------------------
// Fused local (causal, position-biased) + memory attention + sigmoid gate.
// grid = (T/16, B*H), block = 32 (one wave per 16-query-row tile).
// ---------------------------------------------------------------------------
__global__ void __launch_bounds__(32)
attn_fused(const __bf16* __restrict__ Q, const __bf16* __restrict__ Kl,
           const __bf16* __restrict__ VTl, const __bf16* __restrict__ KM,
           const __bf16* __restrict__ VMT, const float* __restrict__ pos,
           const float* __restrict__ memBias, __bf16* __restrict__ attnOut,
           int T, int Bsz, int Mlen) {
  __shared__ __align__(32) __bf16 pTile[16 * 32];
  const int lane = threadIdx.x;
  const int ln = lane & 15;
  const int half = lane >> 4;
  const int t0 = blockIdx.x * 16;
  const int bh = blockIdx.y;              // == b*NHEAD + h
  const int b = bh / NHEAD;
  const int h = bh % NHEAD;

  // Q fragments for K-steps 0 (d 0..31) and 1 (d 32..63); A-layout per lane:
  // row = ln, K = kbase..kbase+15, kbase = half*16.
  const __bf16* qrow = Q + ((size_t)bh * T + t0 + ln) * DH;
  v16bf qf0 = cat8(*(const v8bf*)(qrow + half * 16),
                   *(const v8bf*)(qrow + half * 16 + 8));
  v16bf qf1 = cat8(*(const v8bf*)(qrow + 32 + half * 16),
                   *(const v8bf*)(qrow + 32 + half * 16 + 8));

  float oloc[32], omem[32];
  attn_pass<true>(qf0, qf1, Kl + (size_t)bh * T * DH,
                  VTl + (size_t)bh * DH * T, T, t0 + 16,
                  pos + (size_t)h * T * T, (long)T, t0, lane, pTile, oloc);
  attn_pass<false>(qf0, qf1, KM + (size_t)bh * Mlen * DH,
                   VMT + (size_t)bh * DH * Mlen, Mlen, Mlen, nullptr, 0, t0,
                   lane, pTile, omem);

  const float g = 1.f / (1.f + __expf(-memBias[h]));
#pragma unroll
  for (int dt = 0; dt < 4; ++dt)
#pragma unroll
    for (int r = 0; r < 8; ++r) {
      const int t = t0 + r + 8 * half;
      const int e = h * DH + dt * 16 + ln;
      const float val = (1.f - g) * oloc[dt * 8 + r] + g * omem[dt * 8 + r];
      attnOut[((size_t)t * Bsz + b) * E_DIM + e] = (__bf16)val;
    }
}

// ---------------------------------------------------------------------------
extern "C" void kernel_launch(void* const* d_in, const int* in_sizes, int n_in,
                              void* d_out, int out_size, void* d_ws,
                              size_t ws_size, hipStream_t stream) {
  const int T = 2048, B = 2, M = 1024;
  const float scaling = 0.125f;  // D^-0.5, D=64

  const float* query = (const float*)d_in[0];
  const float* pos = (const float*)d_in[1];
  const float* lcr = (const float*)d_in[2];  // [2, M, B, E]
  const float* q_w = (const float*)d_in[3];
  const float* q_b = (const float*)d_in[4];
  const float* k_w = (const float*)d_in[5];
  const float* k_b = (const float*)d_in[6];
  const float* v_w = (const float*)d_in[7];
  const float* v_b = (const float*)d_in[8];
  const float* out_w = (const float*)d_in[9];
  const float* out_b = (const float*)d_in[10];
  const float* mem_bias = (const float*)d_in[11];

  char* ws = (char*)d_ws;
  size_t off = 0;
  auto alloc = [&](size_t bytes) {
    void* p = ws + off;
    off += (bytes + 255) & ~(size_t)255;
    return p;
  };

  const size_t nTB = (size_t)T * B * E_DIM;   // 4M elems
  const size_t nMB = (size_t)M * B * E_DIM;   // 2M elems
  const size_t nW = (size_t)E_DIM * E_DIM;    // 1M elems

  __bf16* qry_bf = (__bf16*)alloc(nTB * 2);
  __bf16* wq = (__bf16*)alloc(nW * 2);
  __bf16* wk = (__bf16*)alloc(nW * 2);
  __bf16* wv = (__bf16*)alloc(nW * 2);
  __bf16* wo = (__bf16*)alloc(nW * 2);
  __bf16* mkin = (__bf16*)alloc(nMB * 2);
  __bf16* mvin = (__bf16*)alloc(nMB * 2);
  __bf16* Qp = (__bf16*)alloc(nTB * 2);       // [B,H,T,D]
  __bf16* Kp = (__bf16*)alloc(nTB * 2);       // [B,H,T,D]
  __bf16* VTp = (__bf16*)alloc(nTB * 2);      // [B,H,D,T]
  __bf16* KMp = (__bf16*)alloc(nMB * 2);      // [B,H,M,D]
  __bf16* VMTp = (__bf16*)alloc(nMB * 2);     // [B,H,D,M]
  __bf16* attnOut = (__bf16*)alloc(nTB * 2);  // [T*B, E]

  auto cvt = [&](const float* in, __bf16* out, long n) {
    int blocks = (int)((n + 1023) / 1024);
    if (blocks > 4096) blocks = 4096;
    cvt_f32_bf16<<<blocks, 256, 0, stream>>>(in, out, n);
  };
  cvt(query, qry_bf, (long)nTB);
  cvt(q_w, wq, (long)nW);
  cvt(k_w, wk, (long)nW);
  cvt(v_w, wv, (long)nW);
  cvt(out_w, wo, (long)nW);
  cvt(lcr, mkin, (long)nMB);
  cvt(lcr + nMB, mvin, (long)nMB);

  const dim3 blk(128);
  {  // projections from query: rows = T*B = 4096
    dim3 grd(E_DIM / 64, (T * B) / 64);
    gemm_nt_bf16<<<grd, blk, 0, stream>>>(qry_bf, wq, q_b, Qp, T * B, B, 0,
                                          scaling);
    gemm_nt_bf16<<<grd, blk, 0, stream>>>(qry_bf, wk, k_b, Kp, T * B, B, 0,
                                          1.0f);
    gemm_nt_bf16<<<grd, blk, 0, stream>>>(qry_bf, wv, v_b, VTp, T * B, B, 1,
                                          1.0f);
  }
  {  // memory projections: rows = M*B = 2048
    dim3 grd(E_DIM / 64, (M * B) / 64);
    gemm_nt_bf16<<<grd, blk, 0, stream>>>(mkin, wk, k_b, KMp, M * B, B, 0,
                                          1.0f);
    gemm_nt_bf16<<<grd, blk, 0, stream>>>(mvin, wv, v_b, VMTp, M * B, B, 1,
                                          1.0f);
  }

  attn_fused<<<dim3(T / 16, B * NHEAD), 32, 0, stream>>>(
      Qp, Kp, VTp, KMp, VMTp, pos, mem_bias, attnOut, T, B, M);

  {  // output projection, f32 result straight into d_out
    dim3 grd(E_DIM / 64, (T * B) / 64);
    gemm_nt_bf16<<<grd, blk, 0, stream>>>(attnOut, wo, out_b, (float*)d_out,
                                          T * B, B, 2, 1.0f);
  }
  (void)in_sizes; (void)n_in; (void)out_size; (void)ws_size;
}
